// GlobalAttention_4226247819437
// MI455X (gfx1250) — matrix operand order repaired and checked by reading, not compiled, not run on hardware
//
#include <hip/hip_runtime.h>
#include <math.h>

#define B_   4
#define C_   256
#define CQK  32
#define N_   4096

typedef __attribute__((ext_vector_type(16))) __bf16 v16bf;
typedef __attribute__((ext_vector_type(8)))  float  v8f;
typedef __attribute__((ext_vector_type(4)))  float  f32x4;
typedef unsigned short u16;
typedef unsigned int   u32;

union BV16 {
  v16bf v;
  f32x4 f4[2];
  u16   u[16];
};

union PK8 {        // 8 packed bf16 = one 16-byte store
  f32x4 f;
  u16   u[8];
};

__device__ __forceinline__ u16 f2bf(float f) {
  __bf16 h = (__bf16)f;                 // native cvt, round-to-nearest-even
  return __builtin_bit_cast(u16, h);
}

// ---------------------------------------------------------------------------
// Kernel 0: transpose + convert  x[b][c][n] f32  ->  xT[b][n][c] bf16.
// 32x32 tiles through LDS so both global phases are fully coalesced.
// grid (N_/32, C_/32, B_), block (32, 8).
// ---------------------------------------------------------------------------
__global__ void __launch_bounds__(256)
transpose_kernel(const float* __restrict__ x, u16* __restrict__ xT) {
  __shared__ u16 t[32][33];
  const int b  = blockIdx.z;
  const int n0 = blockIdx.x * 32;
  const int c0 = blockIdx.y * 32;
  const int tx = threadIdx.x, ty = threadIdx.y;
#pragma unroll
  for (int i = 0; i < 4; ++i) {
    const int c = c0 + ty + i * 8;
    t[ty + i * 8][tx] = f2bf(x[((size_t)b * C_ + c) * N_ + n0 + tx]);
  }
  __syncthreads();
#pragma unroll
  for (int i = 0; i < 4; ++i) {
    const int n = n0 + ty + i * 8;
    xT[((size_t)b * N_ + n) * C_ + c0 + tx] = t[tx][ty + i * 8];
  }
}

// ---------------------------------------------------------------------------
// Kernel 1: all three projections as WMMA GEMMs.  One wave per block computes
// a 16(out-ch) x 64(pixel) tile of  W[oc,c] @ xT[n,c]^T, K fully unrolled
// (8 x 32) so the scheduler can hoist B-tile loads across WMMAs.
//   blockIdx.y: 0-1 -> q tile, 2-3 -> k tile, 4-19 -> v tile.
// Outputs: q,k as bf16 [b][n][32] (packed 16B stores); v transposed vt[b][c][m].
// grid (N_/64, 20, B_), block 32.
// ---------------------------------------------------------------------------
__global__ void __launch_bounds__(32)
proj_gemm_kernel(const u16* __restrict__ xT,
                 const float* __restrict__ Wq, const float* __restrict__ bq,
                 const float* __restrict__ Wk, const float* __restrict__ bk,
                 const float* __restrict__ Wv, const float* __restrict__ bv,
                 u16* __restrict__ qb, u16* __restrict__ kb,
                 u16* __restrict__ vb) {
  const int lane = threadIdx.x;
  const int col  = lane & 15;
  const int hi   = lane >> 4;
  const int b    = blockIdx.z;
  const int n0   = blockIdx.x * 64;
  const int y    = blockIdx.y;

  const float* W;  const float* bias;  int oc0;  int mode;
  if (y < 2)      { W = Wq; bias = bq; oc0 = y * 16;        mode = 0; }
  else if (y < 4) { W = Wk; bias = bk; oc0 = (y - 2) * 16;  mode = 1; }
  else            { W = Wv; bias = bv; oc0 = (y - 4) * 16;  mode = 2; }

  v8f D[4];
#pragma unroll
  for (int s = 0; s < 4; ++s)
#pragma unroll
    for (int r = 0; r < 8; ++r) D[s][r] = 0.f;

  const u16* xbase = xT + (size_t)b * N_ * C_;

#pragma unroll
  for (int kc = 0; kc < C_; kc += 32) {
    // A operand: W tile (16 oc x 32 c), f32 -> bf16 on the fly.
    // A layout: lane row = L%16; K = kc + hi*8 + e  and  kc + 16 + hi*8 + e.
    BV16 A;
    {
      const float* wr = W + (size_t)(oc0 + col) * C_ + kc;
#pragma unroll
      for (int e = 0; e < 8; ++e) {
        A.u[e]     = f2bf(wr[hi * 8 + e]);
        A.u[8 + e] = f2bf(wr[16 + hi * 8 + e]);
      }
    }
    // B operands: xT rows, K contiguous.  4 pixel sub-tiles.
#pragma unroll
    for (int s = 0; s < 4; ++s) {
      const u16* xr = xbase + (size_t)(n0 + s * 16 + col) * C_ + kc + hi * 8;
      BV16 Bx;
      Bx.f4[0] = *(const f32x4*)(xr);
      Bx.f4[1] = *(const f32x4*)(xr + 16);
      D[s] = __builtin_amdgcn_wmma_f32_16x16x32_bf16(false, A.v, false, Bx.v,
                                                     (short)0, D[s], false, false);
    }
  }

  // bias + store.  D layout: lane col = pixel, reg r = oc r + 8*hi.
  if (mode == 2) {
#pragma unroll
    for (int s = 0; s < 4; ++s) {
      const int n = n0 + s * 16 + col;
#pragma unroll
      for (int r = 0; r < 8; ++r) {
        const int oc = oc0 + r + 8 * hi;
        vb[(size_t)(b * C_ + oc) * N_ + n] = f2bf(D[s][r] + bias[oc]);
      }
    }
  } else {
    u16* dst = (mode == 0) ? qb : kb;
#pragma unroll
    for (int s = 0; s < 4; ++s) {
      const int n = n0 + s * 16 + col;
      PK8 pk;
#pragma unroll
      for (int r = 0; r < 8; ++r)
        pk.u[r] = f2bf(D[s][r] + bias[oc0 + 8 * hi + r]);
      *(f32x4*)&dst[(size_t)(b * N_ + n) * CQK + oc0 + 8 * hi] = pk.f;
    }
  }
}

// ---------------------------------------------------------------------------
// Kernel 2: fused flash-attention. One wave per block handles 16 query rows x
// 128 output channels, streaming key tiles of 64:
//   S(16x64)  = Q(16x32) @ K^T          -> 4x v_wmma_f32_16x16x32_bf16
//   online softmax (row stats via shfl within 16-lane halves)
//   O(16x128) += P(16x64) @ V(64x128)   -> 16x v_wmma_f32_16x16x32_bf16
// Epilogue stages O through LDS so out/x traffic is 64B-contiguous per lane.
// grid (B_*N_/16, 2), block 32.
// ---------------------------------------------------------------------------
__global__ void __launch_bounds__(32)
attn_kernel(const u16* __restrict__ qb, const u16* __restrict__ kb,
            const u16* __restrict__ vb, const float* __restrict__ x,
            float* __restrict__ out) {
  __shared__ __align__(16) u16   ldsP[16 * 64];   // P tile, bf16
  __shared__ __align__(16) float Ot[128][20];     // epilogue staging (padded)

  const int lane = threadIdx.x;
  const int col  = lane & 15;
  const int hi   = lane >> 4;
  const int b    = blockIdx.x >> 8;              // 256 row tiles per batch
  const int n0   = (blockIdx.x & 255) << 4;
  const int c0   = blockIdx.y << 7;              // 128-channel chunk

  // Q tile (A operand, resident).
  BV16 A;
  {
    const u16* qrow = qb + (size_t)(b * N_ + n0 + col) * CQK;
    A.f4[0] = *(const f32x4*)(qrow + hi * 8);
    A.f4[1] = *(const f32x4*)(qrow + 16 + hi * 8);
  }

  v8f O[8];
#pragma unroll
  for (int s = 0; s < 8; ++s)
#pragma unroll
    for (int r = 0; r < 8; ++r) O[s][r] = 0.f;

  float m_i[16], l_i[16];
#pragma unroll
  for (int i = 0; i < 16; ++i) { m_i[i] = -INFINITY; l_i[i] = 0.f; }

  const u16* kbase = kb + (size_t)b * N_ * CQK;
  const u16* vbase = vb + (size_t)b * C_ * N_;

  v8f z;
#pragma unroll
  for (int r = 0; r < 8; ++r) z[r] = 0.f;

  for (int mb = 0; mb < N_; mb += 64) {
    // ---- scores: S(16x64) as 4 sub-tiles ----------------------------------
    v8f S[4];
#pragma unroll
    for (int t = 0; t < 4; ++t) {
      const u16* krow = kbase + (size_t)(mb + t * 16 + col) * CQK + hi * 8;
      BV16 Bk;
      Bk.f4[0] = *(const f32x4*)(krow);
      Bk.f4[1] = *(const f32x4*)(krow + 16);
      S[t] = __builtin_amdgcn_wmma_f32_16x16x32_bf16(false, A.v, false, Bk.v,
                                                     (short)0, z, false, false);
    }
    if (mb + 64 < N_) {
      __builtin_prefetch(kbase + (size_t)(mb + 64 + col) * CQK, 0, 1);
      __builtin_prefetch(vbase + (size_t)(c0 + col) * N_ + mb + 64, 0, 1);
    }

    // ---- online softmax row stats -----------------------------------------
    float nm[16];
#pragma unroll
    for (int r = 0; r < 8; ++r) {
      float v = fmaxf(fmaxf(S[0][r], S[1][r]), fmaxf(S[2][r], S[3][r]));
      v = fmaxf(v, __shfl_xor(v, 1, 32));
      v = fmaxf(v, __shfl_xor(v, 2, 32));
      v = fmaxf(v, __shfl_xor(v, 4, 32));
      v = fmaxf(v, __shfl_xor(v, 8, 32));
      nm[r]     = __shfl(v, 0, 32);
      nm[r + 8] = __shfl(v, 16, 32);
    }
    float alpha[16];
#pragma unroll
    for (int i = 0; i < 16; ++i) {
      float mnew = fmaxf(m_i[i], nm[i]);
      alpha[i] = __expf(m_i[i] - mnew);
      m_i[i] = mnew;
      l_i[i] *= alpha[i];
    }
#pragma unroll
    for (int r = 0; r < 8; ++r) {
      const float ar = hi ? alpha[r + 8] : alpha[r];
#pragma unroll
      for (int s = 0; s < 8; ++s) O[s][r] *= ar;
    }

    // ---- P = exp(S - m), row sums, bf16 -> LDS ----------------------------
#pragma unroll
    for (int r = 0; r < 8; ++r) {
      const float mrow = hi ? m_i[r + 8] : m_i[r];
      float rs = 0.f;
#pragma unroll
      for (int t = 0; t < 4; ++t) {
        float p = __expf(S[t][r] - mrow);
        ldsP[(r + 8 * hi) * 64 + t * 16 + col] = f2bf(p);
        rs += p;
      }
      rs += __shfl_xor(rs, 1, 32);
      rs += __shfl_xor(rs, 2, 32);
      rs += __shfl_xor(rs, 4, 32);
      rs += __shfl_xor(rs, 8, 32);
      l_i[r]     += __shfl(rs, 0, 32);
      l_i[r + 8] += __shfl(rs, 16, 32);
    }
    asm volatile("s_wait_dscnt 0x0" ::: "memory");

    // reload P in WMMA A layout: two 16x32 halves of the 16x64 tile
    BV16 Pa0, Pa1;
    {
      const u16* prow = &ldsP[col * 64];
      Pa0.f4[0] = *(const f32x4*)(prow + hi * 16);
      Pa0.f4[1] = *(const f32x4*)(prow + 8 + hi * 16);
      Pa1.f4[0] = *(const f32x4*)(prow + 32 + hi * 16);
      Pa1.f4[1] = *(const f32x4*)(prow + 40 + hi * 16);
    }

    // ---- O += P @ V : all Pa0 tiles first, then all Pa1 (no back-to-back
    // same-accumulator WMMA hazards) ----------------------------------------
#pragma unroll
    for (int s = 0; s < 8; ++s) {
      const u16* vrow = vbase + (size_t)(c0 + s * 16 + col) * N_ + mb + hi * 16;
      BV16 Bv;
      Bv.f4[0] = *(const f32x4*)(vrow);
      Bv.f4[1] = *(const f32x4*)(vrow + 8);
      O[s] = __builtin_amdgcn_wmma_f32_16x16x32_bf16(false, Pa0.v, false, Bv.v,
                                                     (short)0, O[s], false, false);
    }
#pragma unroll
    for (int s = 0; s < 8; ++s) {
      const u16* vrow = vbase + (size_t)(c0 + s * 16 + col) * N_ + mb + 32 + hi * 16;
      BV16 Bv;
      Bv.f4[0] = *(const f32x4*)(vrow);
      Bv.f4[1] = *(const f32x4*)(vrow + 8);
      O[s] = __builtin_amdgcn_wmma_f32_16x16x32_bf16(false, Pa1.v, false, Bv.v,
                                                     (short)0, O[s], false, false);
    }
  }

  // ---- finalize: scale, stage through LDS, contiguous 64B out/x traffic ---
#pragma unroll
  for (int i = 0; i < 16; ++i) l_i[i] = 1.f / l_i[i];
#pragma unroll
  for (int s = 0; s < 8; ++s) {
    const float* li = l_i;
#pragma unroll
    for (int r = 0; r < 8; ++r)
      Ot[s * 16 + col][r + 8 * hi] = O[s][r] * (hi ? li[r + 8] : li[r]);
  }
  asm volatile("s_wait_dscnt 0x0" ::: "memory");

#pragma unroll
  for (int j = 0; j < 4; ++j) {
    const int cc = lane + 32 * j;          // 0..127 local channel
    const int c  = c0 + cc;
    const float* xo = x + (size_t)(b * C_ + c) * N_ + n0;
    float* oo = out + (size_t)(b * C_ + c) * N_ + n0;
#pragma unroll
    for (int g = 0; g < 4; ++g) {
      f32x4 ov = *(const f32x4*)&Ot[cc][g * 4];
      f32x4 xv = *(const f32x4*)(xo + g * 4);
      ov += xv;
      *(f32x4*)(oo + g * 4) = ov;
    }
  }
}

// ---------------------------------------------------------------------------
extern "C" void kernel_launch(void* const* d_in, const int* in_sizes, int n_in,
                              void* d_out, int out_size, void* d_ws, size_t ws_size,
                              hipStream_t stream) {
  const float* x  = (const float*)d_in[0];
  const float* Wq = (const float*)d_in[1];
  const float* bq = (const float*)d_in[2];
  const float* Wk = (const float*)d_in[3];
  const float* bk = (const float*)d_in[4];
  const float* Wv = (const float*)d_in[5];
  const float* bv = (const float*)d_in[6];
  float* out = (float*)d_out;

  u16* qb = (u16*)d_ws;                                    //  1 MB  q [b][n][32]
  u16* kb = qb + (size_t)B_ * N_ * CQK;                    //  1 MB  k [b][n][32]
  u16* vb = kb + (size_t)B_ * N_ * CQK;                    //  8 MB  vt[b][c][m]
  u16* xT = vb + (size_t)B_ * C_ * N_;                     //  8 MB  xT[b][n][c]

  transpose_kernel<<<dim3(N_ / 32, C_ / 32, B_), dim3(32, 8), 0, stream>>>(x, xT);
  proj_gemm_kernel<<<dim3(N_ / 64, 20, B_), 32, 0, stream>>>(
      xT, Wq, bq, Wk, bk, Wv, bv, qb, kb, vb);
  attn_kernel<<<dim3(B_ * (N_ / 16), 2), 32, 0, stream>>>(qb, kb, vb, x, out);
}
